// SpatialCrossAttention_82978768159185
// MI455X (gfx1250) — matrix operand order repaired and checked by reading, not compiled, not run on hardware
//
#include <hip/hip_runtime.h>

// ---------------- problem constants ----------------
constexpr int Bb = 4;
constexpr int CM = 256;
constexpr int CT = 64;
constexpr int CA = 128;
constexpr int H  = 80;
constexpr int W  = 80;
constexpr int N  = H * W;           // 6400 tokens
constexpr float SCALE = 0.08838834764831845f;  // CA^-0.5

typedef __bf16 bf16_t;
typedef __attribute__((ext_vector_type(16))) __bf16 v16bf;
typedef __attribute__((ext_vector_type(8)))  __bf16 v8bf;
typedef __attribute__((ext_vector_type(8)))  float  v8f;

// ---------------- helpers ----------------
__device__ __forceinline__ bf16_t f2bf(float f) {
    unsigned u = __float_as_uint(f);
    u += 0x7FFFu + ((u >> 16) & 1u);           // round-to-nearest-even
    unsigned short h = (unsigned short)(u >> 16);
    bf16_t r;
    __builtin_memcpy(&r, &h, 2);
    return r;
}

// A fragment (16x32 bf16): lane holds row (lane&15); K run = [half*16 .. +16)
// (contiguous order, matches B fragments)
__device__ __forceinline__ v16bf load_a_frag(const bf16_t* row, int half) {
    v8bf lo = *(const v8bf*)(row + half * 16);
    v8bf hi = *(const v8bf*)(row + half * 16 + 8);
    return __builtin_shufflevector(lo, hi, 0,1,2,3,4,5,6,7,8,9,10,11,12,13,14,15);
}

// B fragment (32x16 bf16): lane holds column (lane&15); K run = (half*16 .. +16)
// column-contiguous storage -> one 32B run
__device__ __forceinline__ v16bf load_b_frag(const bf16_t* p) {
    v8bf lo = *(const v8bf*)p;
    v8bf hi = *(const v8bf*)(p + 8);
    return __builtin_shufflevector(lo, hi, 0,1,2,3,4,5,6,7,8,9,10,11,12,13,14,15);
}

__device__ __forceinline__ v8f wmma_bf16(v16bf a, v16bf b, v8f c) {
    return __builtin_amdgcn_wmma_f32_16x16x32_bf16(
        false, a, false, b, (short)0, c, false, false);
}

// broadcast from lane (byteAddr/4); byteAddr = half*32 + 4*r folds 4*r into DS offset
__device__ __forceinline__ float lane_bcast(float v, int byteAddr) {
    return __int_as_float(__builtin_amdgcn_ds_bpermute(byteAddr, __float_as_int(v)));
}

// ---------------- kernel 1: Q/K/V projection (1x1 conv) ----------------
// Q,K -> [B][N][CA] bf16 (token-major, channel contiguous); V -> [B][CA][N] bf16
__global__ void __launch_bounds__(CA)
qkv_proj(const float* __restrict__ met, const float* __restrict__ ter,
         const float* __restrict__ wq, const float* __restrict__ bq,
         const float* __restrict__ wk, const float* __restrict__ bk,
         const float* __restrict__ wv, const float* __restrict__ bv,
         bf16_t* __restrict__ Qb, bf16_t* __restrict__ Kb,
         bf16_t* __restrict__ Vb) {
    const int o = threadIdx.x;      // 0..CA-1
    const int n = blockIdx.x;       // token
    const int b = blockIdx.y;       // batch
    const float* metc = met + (size_t)b * CM * N + n;   // uniform addr -> scalar loads
    const float* terc = ter + (size_t)b * CT * N + n;
    const float* wqr = wq + o * CM;
    const float* wkr = wk + o * CT;
    const float* wvr = wv + o * CT;
    float accq = bq[o], acck = bk[o], accv = bv[o];
#pragma unroll 8
    for (int c = 0; c < CM; ++c)
        accq = fmaf(wqr[c], metc[(size_t)c * N], accq);
#pragma unroll 8
    for (int c = 0; c < CT; ++c) {
        float t = terc[(size_t)c * N];
        acck = fmaf(wkr[c], t, acck);
        accv = fmaf(wvr[c], t, accv);
    }
    Qb[((size_t)(b * N + n)) * CA + o] = f2bf(accq * SCALE);  // fold softmax scale into Q
    Kb[((size_t)(b * N + n)) * CA + o] = f2bf(acck);
    Vb[((size_t)(b * CA + o)) * N + n] = f2bf(accv);
}

// ---------------- kernel 2: wo -> bf16 ----------------
__global__ void cvt_wo(const float* __restrict__ wo, bf16_t* __restrict__ wob) {
    int i = blockIdx.x * 256 + threadIdx.x;
    if (i < CM * CA) wob[i] = f2bf(wo[i]);
}

// ---------------- kernel 3: fused flash attention (transposed-S trick) ----------------
// 1 wave owns a 16-query tile and the full CA=128 output; 64 keys per iteration.
// S' = K x Q^T so each lane holds its own query's P row along j:
// PV A-fragments are built fully in-lane (no LDS transpose).
__global__ void __launch_bounds__(128, 1) __attribute__((amdgpu_waves_per_eu(1, 2)))
attn_flash(const bf16_t* __restrict__ Qb, const bf16_t* __restrict__ Kb,
           const bf16_t* __restrict__ Vb, float* __restrict__ Obuf) {
    const int lane = threadIdx.x & 31;
    const int wv_  = threadIdx.x >> 5;       // wave in block
    const int half = lane >> 4;
    const int lm   = lane & 15;
    const int bbase = half * 32;             // bpermute byte base (lane 8*half)

    const int blkPerB = N / 64;
    const int b  = blockIdx.x / blkPerB;
    const int i0 = (blockIdx.x % blkPerB) * 64 + wv_ * 16;

    // resident Q as B fragments: column i = i0+lm, contraction c
    const bf16_t* Qrow = Qb + ((size_t)(b * N + i0 + lm)) * CA + half * 16;
    v16bf qf[4];
#pragma unroll
    for (int kc = 0; kc < 4; ++kc)
        qf[kc] = load_b_frag(Qrow + kc * 32);

    v8f Oacc[8];
#pragma unroll
    for (int cb = 0; cb < 8; ++cb) Oacc[cb] = (v8f){0.f,0.f,0.f,0.f,0.f,0.f,0.f,0.f};
    float m = -1e30f;   // per-lane softmax state for query i0+lm (both halves identical)
    float l = 0.f;

    const bf16_t* Kbase = Kb + (size_t)(b * N) * CA;
    const bf16_t* Vbase = Vb + (size_t)(b * CA) * N;

#pragma unroll 1
    for (int j0 = 0; j0 < N; j0 += 64) {
        // ---- S' = K x Q^T over 64 keys: 4 16x16 tiles ----
        v8f Sa = (v8f){0.f,0.f,0.f,0.f,0.f,0.f,0.f,0.f};
        v8f Sb = Sa, Sc = Sa, Sd = Sa;
        const bf16_t* ka = Kbase + (size_t)(j0 + lm) * CA;        // A rows = keys
        const bf16_t* kb = Kbase + (size_t)(j0 + 16 + lm) * CA;
        const bf16_t* kc_ = Kbase + (size_t)(j0 + 32 + lm) * CA;
        const bf16_t* kd = Kbase + (size_t)(j0 + 48 + lm) * CA;
#pragma unroll
        for (int kc = 0; kc < 4; ++kc) {
            Sa = wmma_bf16(load_a_frag(ka + kc * 32, half), qf[kc], Sa);
            Sb = wmma_bf16(load_a_frag(kb + kc * 32, half), qf[kc], Sb);
            Sc = wmma_bf16(load_a_frag(kc_ + kc * 32, half), qf[kc], Sc);
            Sd = wmma_bf16(load_a_frag(kd + kc * 32, half), qf[kc], Sd);
        }
        // pull next j-block's K rows toward the WGP while we do softmax
        __builtin_prefetch(ka + (size_t)64 * CA, 0, 3);
        __builtin_prefetch(kb + (size_t)64 * CA, 0, 3);
        __builtin_prefetch(kc_ + (size_t)64 * CA, 0, 3);
        __builtin_prefetch(kd + (size_t)64 * CA, 0, 3);

        // ---- online softmax; per-lane query = lm, j spread over 4 tiles x 2 halves ----
        float t = fmaxf(fmaxf(Sa[0], Sb[0]), fmaxf(Sc[0], Sd[0]));
#pragma unroll
        for (int r = 1; r < 8; ++r)
            t = fmaxf(t, fmaxf(fmaxf(Sa[r], Sb[r]), fmaxf(Sc[r], Sd[r])));
        t = fmaxf(t, __shfl_xor(t, 16, 32));          // other half's j values
        float mn = fmaxf(m, t);
        float cf = __expf(m - mn);
        m = mn;
        float s = 0.f;
#pragma unroll
        for (int r = 0; r < 8; ++r) {
            Sa[r] = __expf(Sa[r] - mn);
            Sb[r] = __expf(Sb[r] - mn);
            Sc[r] = __expf(Sc[r] - mn);
            Sd[r] = __expf(Sd[r] - mn);
            s += (Sa[r] + Sb[r]) + (Sc[r] + Sd[r]);
        }
        s += __shfl_xor(s, 16, 32);
        l = l * cf + s;

        // ---- P A-fragments, fully in-lane ----
        // pf0: elements 0..7 <-> j = j0+8*half.., 8..15 <-> j = j0+16+8*half..
        // pf1: same for j0+32 / j0+48
        v16bf pf0, pf1;
#pragma unroll
        for (int r = 0; r < 8; ++r) {
            pf0[r]     = f2bf(Sa[r]);
            pf0[8 + r] = f2bf(Sb[r]);
            pf1[r]     = f2bf(Sc[r]);
            pf1[8 + r] = f2bf(Sd[r]);
        }

        // ---- rescale O: row i = r + 8*half needs cf of that query (lane i&15) ----
        float cfr[8];
#pragma unroll
        for (int r = 0; r < 8; ++r) cfr[r] = lane_bcast(cf, bbase + 4 * r);
#pragma unroll
        for (int cb = 0; cb < 8; ++cb)
#pragma unroll
            for (int r = 0; r < 8; ++r) Oacc[cb][r] *= cfr[r];

        // ---- O += P x V^T over 8 channel tiles, K=64 -> 2 WMMA each ----
#pragma unroll
        for (int cb = 0; cb < 8; ++cb) {
            const bf16_t* vp = Vbase + (size_t)(cb * 16 + lm) * N + j0 + half * 8;
            Oacc[cb] = wmma_bf16(pf0, __builtin_shufflevector(*(const v8bf*)vp, *(const v8bf*)(vp + 16), 0,1,2,3,4,5,6,7,8,9,10,11,12,13,14,15), Oacc[cb]);
            Oacc[cb] = wmma_bf16(pf1, __builtin_shufflevector(*(const v8bf*)(vp + 32), *(const v8bf*)(vp + 48), 0,1,2,3,4,5,6,7,8,9,10,11,12,13,14,15), Oacc[cb]);
        }
    }

    // ---- normalize rows and write O as [B][N][CA] f32 ----
    float inv = 1.f / l;
    float invr[8];
#pragma unroll
    for (int r = 0; r < 8; ++r) invr[r] = lane_bcast(inv, bbase + 4 * r);
    float* Op = Obuf + ((size_t)(b * N + i0)) * CA;
#pragma unroll
    for (int cb = 0; cb < 8; ++cb)
#pragma unroll
        for (int r = 0; r < 8; ++r) {
            int row = r + 8 * half;
            Op[(size_t)row * CA + cb * 16 + lm] = Oacc[cb][r] * invr[r];
        }
}

// ---------------- kernel 4: out = met + wo x O + bo (WMMA) ----------------
// 1 wave per 16(o) x 16(n) tile; blockDim 256 (8 waves).
__global__ void __launch_bounds__(256, 1)
out_proj(const float* __restrict__ met, const bf16_t* __restrict__ wob,
         const float* __restrict__ bo, const float* __restrict__ Obuf,
         float* __restrict__ out) {
    const int lane = threadIdx.x & 31;
    const int wv_  = threadIdx.x >> 5;
    const int half = lane >> 4;
    const int lm   = lane & 15;

    const int nTiles = N / 16;                     // 400
    int t = blockIdx.x * 8 + wv_;
    const int b   = t / (16 * nTiles);
    const int rem = t % (16 * nTiles);
    const int ob  = rem / nTiles;                  // o tile
    const int nb  = rem % nTiles;                  // n tile

    const bf16_t* arow = wob + (size_t)(ob * 16 + lm) * CA;
    const float*  brow = Obuf + ((size_t)(b * N) + nb * 16 + lm) * CA;

    v8f acc = (v8f){0.f,0.f,0.f,0.f,0.f,0.f,0.f,0.f};
#pragma unroll
    for (int kc = 0; kc < 4; ++kc) {
        v16bf af = load_a_frag(arow + kc * 32, half);
        const float* bp = brow + kc * 32 + half * 16;
        v8f b0 = *(const v8f*)bp;
        v8f b1 = *(const v8f*)(bp + 8);
        v16bf bfv;
#pragma unroll
        for (int i = 0; i < 8; ++i) { bfv[i] = f2bf(b0[i]); bfv[8 + i] = f2bf(b1[i]); }
        acc = wmma_bf16(af, bfv, acc);
    }

#pragma unroll
    for (int r = 0; r < 8; ++r) {
        int row = ob * 16 + r + 8 * half;
        size_t idx = ((size_t)(b * CM + row)) * N + nb * 16 + lm;
        out[idx] = met[idx] + bo[row] + acc[r];
    }
}

// ---------------- launch ----------------
extern "C" void kernel_launch(void* const* d_in, const int* in_sizes, int n_in,
                              void* d_out, int out_size, void* d_ws, size_t ws_size,
                              hipStream_t stream) {
    const float* met = (const float*)d_in[0];
    const float* ter = (const float*)d_in[1];
    const float* wq  = (const float*)d_in[2];
    const float* bq  = (const float*)d_in[3];
    const float* wk  = (const float*)d_in[4];
    const float* bk  = (const float*)d_in[5];
    const float* wv  = (const float*)d_in[6];
    const float* bv  = (const float*)d_in[7];
    const float* wo  = (const float*)d_in[8];
    const float* bo  = (const float*)d_in[9];
    float* out = (float*)d_out;

    char* ws = (char*)d_ws;
    size_t off = 0;
    auto take = [&](size_t bytes) { size_t p = off; off = (off + bytes + 255) & ~(size_t)255; return p; };
    bf16_t* Qb   = (bf16_t*)(ws + take((size_t)Bb * N * CA * 2));
    bf16_t* Kb   = (bf16_t*)(ws + take((size_t)Bb * N * CA * 2));
    bf16_t* Vb   = (bf16_t*)(ws + take((size_t)Bb * CA * N * 2));
    float*  Obuf = (float* )(ws + take((size_t)Bb * N * CA * 4));
    bf16_t* wob  = (bf16_t*)(ws + take((size_t)CM * CA * 2));
    (void)ws_size; (void)in_sizes; (void)n_in; (void)out_size;

    cvt_wo<<<(CM * CA + 255) / 256, 256, 0, stream>>>(wo, wob);
    qkv_proj<<<dim3(N, Bb), CA, 0, stream>>>(met, ter, wq, bq, wk, bk, wv, bv, Qb, Kb, Vb);
    attn_flash<<<Bb * (N / 64), 128, 0, stream>>>(Qb, Kb, Vb, Obuf);
    out_proj<<<(Bb * 16 * (N / 16)) / 8, 256, 0, stream>>>(met, wob, bo, Obuf, out);
}
